// LlamaGQA_69569880260704
// MI455X (gfx1250) — matrix operand order repaired and checked by reading, compiled, not run on hardware
//
#include <hip/hip_runtime.h>
#include <hip/hip_bf16.h>
#include <math.h>
#include <stdint.h>

// LlamaGQA for MI455X / gfx1250 (wave32, WMMA bf16, TDM async tensor loads).
// B=2, T=2048, DIM=4096, H=32, Hkv=8, hd=128.
// Pipeline: proj(Q,+rope) -> proj(K,+rope) -> proj(V, transposed store)
//           -> flash attention (TDM-staged K/V tiles) -> output projection.
// Workspace layout (80 MB total):
//   [0,32MB)   Q  bf16 [b][h][t][128]
//   [32,40MB)  K  bf16 [b][g][t][128]
//   [40,48MB)  Vt bf16 [b][g][128][t]   (transposed for 2D-TDM + contiguous B-frags)
//   [48,80MB)  attn bf16 [b][t][4096]

typedef __bf16 bf16;
typedef __attribute__((ext_vector_type(16))) __bf16 v16bf;
typedef __attribute__((ext_vector_type(8)))  float  v8f;
typedef __attribute__((ext_vector_type(4)))  unsigned int u32x4;
typedef __attribute__((ext_vector_type(8)))  unsigned int u32x8;

#define TDIM 2048
#define DIMC 4096
#define HD   128
#define NQH  32
#define NKVH 8

__device__ __forceinline__ v8f zero8() {
  v8f z = {0.f, 0.f, 0.f, 0.f, 0.f, 0.f, 0.f, 0.f};
  return z;
}

__device__ __forceinline__ v8f wmma_bf16(v16bf a, v16bf b, v8f c) {
  return __builtin_amdgcn_wmma_f32_16x16x32_bf16(
      /*neg_a=*/false, a, /*neg_b=*/false, b,
      /*c_mod=*/(short)0, c, /*reuse_a=*/false, /*reuse_b=*/false);
}

// ---------------------------------------------------------------------------
// Tensor Data Mover: build a D# (groups 0 and 1) for a 2D tile of 8-byte
// elements and issue tensor_load_to_lds. All values are wave-uniform, so the
// compiler materializes them in SGPRs ("s" constraints -> aligned SGPR quads).
//   tile_d0 / tensor_d0 : row length in 8B units
//   tile_d1 / tensor_d1 : number of rows
//   stride0             : row pitch in 8B units
// Completion tracked with TENSORcnt.
// ---------------------------------------------------------------------------
__device__ __forceinline__ void tdm_load_to_lds(unsigned lds_off, const void* gptr,
                                                unsigned tile_d0, unsigned tile_d1,
                                                unsigned tensor_d0, unsigned tensor_d1,
                                                unsigned stride0) {
  unsigned long long ga = (unsigned long long)(size_t)gptr;
  u32x4 g0;
  g0[0] = 1u;                                   // count=1, user descriptor
  g0[1] = lds_off;                              // LDS byte address
  g0[2] = (unsigned)ga;                         // global_addr[31:0]
  g0[3] = (unsigned)(ga >> 32) | (2u << 30);    // global_addr[56:32] | type=2
  u32x8 g1;
  g1[0] = 3u << 16;                             // workgroup_mask=0, data_size=8B
  g1[1] = (tensor_d0 & 0xffffu) << 16;          // tensor_dim0[15:0]
  g1[2] = (tensor_d0 >> 16) | ((tensor_d1 & 0xffffu) << 16); // dim0 hi | dim1 lo
  g1[3] = (tensor_d1 >> 16) | (tile_d0 << 16);  // dim1 hi | tile_dim0
  g1[4] = tile_d1 & 0xffffu;                    // tile_dim1 (tile_dim2=0)
  g1[5] = stride0;                              // tensor_dim0_stride[31:0]
  g1[6] = 0u;                                   // stride0 hi | stride1 lo
  g1[7] = 0u;
  asm volatile("tensor_load_to_lds %0, %1" :: "s"(g0), "s"(g1) : "memory");
}

// A fragment (16x32 bf16): per lane, row m = lane&15; 16 contiguous K elements
// starting at K offset 2*ksel (ksel = (lane>>4)*8).
__device__ __forceinline__ v16bf load_a_f32(const float* __restrict__ p, int ksel) {
  v16bf a;
#pragma unroll
  for (int i = 0; i < 8; ++i) {
    a[i]     = (bf16)p[2 * ksel + i];
    a[i + 8] = (bf16)p[2 * ksel + 8 + i];
  }
  return a;
}
__device__ __forceinline__ v16bf load_a_bf16(const bf16* __restrict__ p, int ksel) {
  v16bf a;
#pragma unroll
  for (int i = 0; i < 8; ++i) {
    a[i]     = p[2 * ksel + i];
    a[i + 8] = p[2 * ksel + 8 + i];
  }
  return a;
}
// B fragment (32x16): per lane, col n = lane&15; 16 contiguous K elements
// starting at K offset (lane>>4)*16.
__device__ __forceinline__ v16bf load_b_f32(const float* __restrict__ p) {
  v16bf b;
#pragma unroll
  for (int i = 0; i < 16; ++i) b[i] = (bf16)p[i];
  return b;
}
__device__ __forceinline__ v16bf load_b_bf16(const bf16* __restrict__ p) {
  v16bf b;
#pragma unroll
  for (int i = 0; i < 16; ++i) b[i] = p[i];
  return b;
}
__device__ __forceinline__ v16bf load_b_lds(const bf16* p) {
  v16bf b;
#pragma unroll
  for (int i = 0; i < 16; ++i) b[i] = p[i];
  return b;
}

// ---------------------------------------------------------------------------
// Fused projection GEMM + RoPE (see round-1 notes; unchanged math).
// mode: 0 = Q (rope, [b,h,t,128]); 1 = K (rope, [b,g,t,128]);
//       2 = V (no rope, transposed [b,g,128,t]).
// ---------------------------------------------------------------------------
__global__ void __launch_bounds__(32)
proj_rope_kernel(const float* __restrict__ x, const float* __restrict__ w,
                 bf16* __restrict__ out, int nheads, int mode)
{
  const int lane = threadIdx.x & 31;
  const int m    = lane & 15;
  const int half = lane >> 4;
  const int ksA  = half * 8;
  const int ksB  = half * 16;
  const int ncol = m;

  const int rowBase = blockIdx.x * 32;
  const int head    = blockIdx.y >> 2;
  const int cpair   = blockIdx.y & 3;
  const int dlo     = cpair * 16 + ncol;     // [0,64)
  const int nLo     = head * HD + dlo;
  const int nHi     = nLo + 64;

  const float* xr0 = x + (size_t)(rowBase + m)      * DIMC;
  const float* xr1 = x + (size_t)(rowBase + 16 + m) * DIMC;
  const float* wLo = w + (size_t)nLo * DIMC;
  const float* wHi = w + (size_t)nHi * DIMC;

  v8f acc00 = zero8(), acc01 = zero8(), acc10 = zero8(), acc11 = zero8();

#pragma unroll 2
  for (int k0 = 0; k0 < DIMC; k0 += 32) {
    __builtin_prefetch(wLo + k0 + 256, 0, 1);
    __builtin_prefetch(wHi + k0 + 256, 0, 1);
    v16bf a0 = load_a_f32(xr0 + k0, ksA);
    v16bf a1 = load_a_f32(xr1 + k0, ksA);
    v16bf b0 = load_b_f32(wLo + k0 + ksB);
    v16bf b1 = load_b_f32(wHi + k0 + ksB);
    acc00 = wmma_bf16(a0, b0, acc00);
    acc01 = wmma_bf16(a0, b1, acc01);
    acc10 = wmma_bf16(a1, b0, acc10);
    acc11 = wmma_bf16(a1, b1, acc11);
  }

  // inv_freq = 10000^(-dlo/64)
  const float freq = expf(-0.14391156516f * (float)dlo);

#pragma unroll
  for (int rt = 0; rt < 2; ++rt) {
    v8f lo = rt ? acc10 : acc00;
    v8f hi = rt ? acc11 : acc01;
    const int rbase = rowBase + rt * 16 + half * 8;
#pragma unroll
    for (int r = 0; r < 8; ++r) {
      const int mg = rbase + r;
      const int bb = mg >> 11;            // / TDIM
      const int tt = mg & (TDIM - 1);
      float vlo = lo[r], vhi = hi[r];
      if (mode < 2) {
        const float th = (float)tt * freq;
        const float sn = sinf(th), cs = cosf(th);
        const float olo = vlo * cs - vhi * sn;   // q*cos + rot_half(q)*sin
        const float ohi = vhi * cs + vlo * sn;
        const size_t base = (((size_t)bb * nheads + head) * TDIM + tt) * HD;
        out[base + dlo]      = (bf16)olo;
        out[base + dlo + 64] = (bf16)ohi;
      } else {
        const size_t base = ((size_t)bb * nheads + head) * HD;
        out[(base + dlo)      * TDIM + tt] = (bf16)vlo;
        out[(base + dlo + 64) * TDIM + tt] = (bf16)vhi;
      }
    }
  }
}

// ---------------------------------------------------------------------------
// Flash attention, one wave per 16 query rows. grid = (T/16, H, B).
// Per 32-key tile: TDM stages K (1D, 8KB contiguous) and V (2D, 128 rows x
// 64B, pitch T*2B) into LDS; s_wait_tensorcnt 1 gates the S=QK^T WMMAs,
// s_wait_tensorcnt 0 gates the O+=PV WMMAs so the V DMA overlaps S math.
// Online softmax in fp32; P crosses C-layout -> A-layout through 1KB LDS.
// ---------------------------------------------------------------------------
__global__ void __launch_bounds__(32)
flash_attn_kernel(const bf16* __restrict__ Q, const bf16* __restrict__ K,
                  const bf16* __restrict__ Vt, bf16* __restrict__ AO)
{
  __shared__ __align__(16) bf16 Klds[32 * HD];   // 8KB: rows kcol..kcol+31
  __shared__ __align__(16) bf16 Vlds[HD * 32];   // 8KB: [d][t-in-tile]
  __shared__ __align__(16) bf16 pbuf[16 * 32];   // 1KB: P transpose staging

  const int lane = threadIdx.x & 31;
  const int m = lane & 15, half = lane >> 4;
  const int ksA = half * 8, ksB = half * 16, ncol = m;

  const int qblk = blockIdx.x, h = blockIdx.y, b = blockIdx.z;
  const int g = h >> 2;                       // kv head (n_rep = 4)

  const bf16* Qp = Q  + (((size_t)(b * NQH  + h)) * TDIM + (size_t)qblk * 16) * HD;
  const bf16* Kp = K  + ((size_t)(b * NKVH + g)) * TDIM * HD;
  const bf16* Vp = Vt + ((size_t)(b * NKVH + g)) * HD * TDIM;

  const unsigned kOff = (unsigned)(size_t)(void*)Klds;
  const unsigned vOff = (unsigned)(size_t)(void*)Vlds;

  v16bf qf[4];
#pragma unroll
  for (int ks = 0; ks < 4; ++ks)
    qf[ks] = load_a_bf16(Qp + (size_t)m * HD + ks * 32, ksA);

  v8f o[8];
#pragma unroll
  for (int c = 0; c < 8; ++c) o[c] = zero8();
  float mstate[8], lstate[8];
#pragma unroll
  for (int r = 0; r < 8; ++r) { mstate[r] = -1e30f; lstate[r] = 0.f; }

  const float scale = 0.08838834764831845f;   // 1/sqrt(128)
  const int kend = qblk * 16 + 16;            // causal limit (exclusive)
  const int nkt  = (kend + 31) >> 5;          // 32-wide key tiles

  for (int kt = 0; kt < nkt; ++kt) {
    const int kcol = kt * 32;

    // Previous iteration's ds reads must land before the DMA overwrites LDS.
    asm volatile("s_wait_dscnt 0x0" ::: "memory");
    // K tile: 32 rows x 128 bf16, contiguous => 1D copy of 1024 x 8B.
    tdm_load_to_lds(kOff, Kp + (size_t)kcol * HD,
                    /*tile_d0=*/1024, /*tile_d1=*/1,
                    /*tensor_d0=*/1024, /*tensor_d1=*/1, /*stride0=*/1024);
    // V tile: 128 rows x 32 bf16 (64B), row pitch T*2B = 512 x 8B.
    tdm_load_to_lds(vOff, Vp + kcol,
                    /*tile_d0=*/8, /*tile_d1=*/HD,
                    /*tensor_d0=*/8, /*tensor_d1=*/HD, /*stride0=*/512);

    __builtin_amdgcn_s_wait_tensorcnt(1);      // K tile resident

    v8f s0 = zero8(), s1 = zero8();
#pragma unroll
    for (int ks = 0; ks < 4; ++ks) {
      v16bf kb0 = load_b_lds(&Klds[(size_t)ncol        * HD + ks * 32 + ksB]);
      v16bf kb1 = load_b_lds(&Klds[(size_t)(16 + ncol) * HD + ks * 32 + ksB]);
      s0 = wmma_bf16(qf[ks], kb0, s0);
      s1 = wmma_bf16(qf[ks], kb1, s1);
    }

    float corr[8];
#pragma unroll
    for (int r = 0; r < 8; ++r) {
      const int qrow = qblk * 16 + half * 8 + r;
      const int c0 = kcol + ncol;
      const int c1 = kcol + 16 + ncol;
      float v0 = (c0 <= qrow) ? s0[r] * scale : -1e30f;
      float v1 = (c1 <= qrow) ? s1[r] * scale : -1e30f;
      float rmax = fmaxf(v0, v1);
      rmax = fmaxf(rmax, __shfl_xor(rmax, 1, 32));
      rmax = fmaxf(rmax, __shfl_xor(rmax, 2, 32));
      rmax = fmaxf(rmax, __shfl_xor(rmax, 4, 32));
      rmax = fmaxf(rmax, __shfl_xor(rmax, 8, 32));
      const float mnew = fmaxf(mstate[r], rmax);
      const float p0 = __expf(v0 - mnew);
      const float p1 = __expf(v1 - mnew);
      float rsum = p0 + p1;
      rsum += __shfl_xor(rsum, 1, 32);
      rsum += __shfl_xor(rsum, 2, 32);
      rsum += __shfl_xor(rsum, 4, 32);
      rsum += __shfl_xor(rsum, 8, 32);
      const float cr = __expf(mstate[r] - mnew);
      lstate[r] = lstate[r] * cr + rsum;
      mstate[r] = mnew;
      corr[r] = cr;
      pbuf[(half * 8 + r) * 32 + ncol]      = (bf16)p0;
      pbuf[(half * 8 + r) * 32 + 16 + ncol] = (bf16)p1;
    }

#pragma unroll
    for (int c = 0; c < 8; ++c)
#pragma unroll
      for (int r = 0; r < 8; ++r) o[c][r] *= corr[r];

    __syncthreads();
    v16bf pf = load_a_bf16(&pbuf[m * 32], ksA);

    __builtin_amdgcn_s_wait_tensorcnt(0);      // V tile resident

#pragma unroll
    for (int c = 0; c < 8; ++c) {
      v16bf vb = load_b_lds(&Vlds[(size_t)(c * 16 + ncol) * 32 + ksB]);
      o[c] = wmma_bf16(pf, vb, o[c]);
    }
  }

  // Store [b][t][h*128 + d] as bf16 for the output projection.
#pragma unroll
  for (int c = 0; c < 8; ++c) {
#pragma unroll
    for (int r = 0; r < 8; ++r) {
      const int tt = qblk * 16 + half * 8 + r;
      const float val = o[c][r] / lstate[r];
      AO[((size_t)b * TDIM + tt) * (NQH * HD) + (size_t)h * HD + c * 16 + ncol] =
          (bf16)val;
    }
  }
}

// ---------------------------------------------------------------------------
// Output projection: Y = attn(bf16) @ wo^T (fp32 weights converted in-register).
// ---------------------------------------------------------------------------
__global__ void __launch_bounds__(32)
out_proj_kernel(const bf16* __restrict__ a, const float* __restrict__ w,
                float* __restrict__ y)
{
  const int lane = threadIdx.x & 31;
  const int m = lane & 15, half = lane >> 4;
  const int ksA = half * 8, ksB = half * 16, ncol = m;
  const int rowBase = blockIdx.x * 32;
  const int colBase = blockIdx.y * 32;

  const bf16* ar0 = a + (size_t)(rowBase + m)      * DIMC;
  const bf16* ar1 = a + (size_t)(rowBase + 16 + m) * DIMC;
  const float* w0 = w + (size_t)(colBase + ncol)      * DIMC;
  const float* w1 = w + (size_t)(colBase + 16 + ncol) * DIMC;

  v8f acc00 = zero8(), acc01 = zero8(), acc10 = zero8(), acc11 = zero8();

#pragma unroll 2
  for (int k0 = 0; k0 < DIMC; k0 += 32) {
    __builtin_prefetch(w0 + k0 + 256, 0, 1);
    __builtin_prefetch(w1 + k0 + 256, 0, 1);
    v16bf a0 = load_a_bf16(ar0 + k0, ksA);
    v16bf a1 = load_a_bf16(ar1 + k0, ksA);
    v16bf b0 = load_b_f32(w0 + k0 + ksB);
    v16bf b1 = load_b_f32(w1 + k0 + ksB);
    acc00 = wmma_bf16(a0, b0, acc00);
    acc01 = wmma_bf16(a0, b1, acc01);
    acc10 = wmma_bf16(a1, b0, acc10);
    acc11 = wmma_bf16(a1, b1, acc11);
  }

#pragma unroll
  for (int rt = 0; rt < 2; ++rt) {
    v8f vlo = rt ? acc10 : acc00;
    v8f vhi = rt ? acc11 : acc01;
    const int rbase = rowBase + rt * 16 + half * 8;
#pragma unroll
    for (int r = 0; r < 8; ++r) {
      y[(size_t)(rbase + r) * DIMC + colBase + ncol]      = vlo[r];
      y[(size_t)(rbase + r) * DIMC + colBase + 16 + ncol] = vhi[r];
    }
  }
}

// ---------------------------------------------------------------------------
extern "C" void kernel_launch(void* const* d_in, const int* in_sizes, int n_in,
                              void* d_out, int out_size, void* d_ws, size_t ws_size,
                              hipStream_t stream) {
  const float* x  = (const float*)d_in[0];
  // d_in[1] = position_ids (arange, recomputed analytically)
  // d_in[2] = causal mask (applied analytically)
  const float* wq = (const float*)d_in[3];
  const float* wk = (const float*)d_in[4];
  const float* wv = (const float*)d_in[5];
  const float* wo = (const float*)d_in[6];
  float* out = (float*)d_out;

  char* ws = (char*)d_ws;
  bf16* Qb  = (bf16*)(ws);                         // 32 MB
  bf16* Kb  = (bf16*)(ws + (size_t)33554432);      //  8 MB
  bf16* Vtb = (bf16*)(ws + (size_t)41943040);      //  8 MB
  bf16* Ab  = (bf16*)(ws + (size_t)50331648);      // 32 MB

  // M = B*T = 4096 rows -> 128 row-blocks of 32.
  proj_rope_kernel<<<dim3(128, NQH * 4),  32, 0, stream>>>(x, wq, Qb,  NQH,  0);
  proj_rope_kernel<<<dim3(128, NKVH * 4), 32, 0, stream>>>(x, wk, Kb,  NKVH, 1);
  proj_rope_kernel<<<dim3(128, NKVH * 4), 32, 0, stream>>>(x, wv, Vtb, NKVH, 2);
  flash_attn_kernel<<<dim3(TDIM / 16, NQH, 2), 32, 0, stream>>>(Qb, Kb, Vtb, Ab);
  out_proj_kernel<<<dim3(128, 128), 32, 0, stream>>>(Ab, wo, out);
}
